// CCRNN_5420248727794
// MI455X (gfx1250) — matrix operand order repaired and checked by reading, not compiled, not run on hardware
//
#include <hip/hip_runtime.h>
#include <hip/hip_bf16.h>

typedef __bf16 bf16;
typedef __attribute__((ext_vector_type(16))) __bf16 v16bf;
typedef __attribute__((ext_vector_type(8)))  __bf16 v8bf;
typedef __attribute__((ext_vector_type(8)))  float  v8f;

#define BB   256
#define TT   40
#define INN  2048
#define EE   512
#define HH   1024
#define VV   5000
#define NPAD 5056   // 79 * 64, >= VV

#define WMMA_BF16(a,b,c) \
  __builtin_amdgcn_wmma_f32_16x16x32_bf16(false,(a),false,(b),(short)0,(c),false,false)

// ---- A fragment loader (contiguous k order, same as the B fragments) ----
// A (16x32 bf16): lane m=lane&15; half=lane>>4: elems 0-15 -> K = k+half*16..
__device__ __forceinline__ v16bf load_a_frag(const bf16* A, int lda, int row, int k, int half) {
  union { v16bf v; v8bf h[2]; } u;
  const bf16* p = A + (size_t)row * lda + k + half * 16;
  u.h[0] = *(const v8bf*)(p);
  u.h[1] = *(const v8bf*)(p + 8);
  return u.v;
}

__device__ __forceinline__ float sigmoidf(float x) { return 1.0f / (1.0f + __expf(-x)); }

// =====================================================================
// Shared GEMM core: WG = 128(M) x 64(N); 8 waves x (16M x 64N each).
// B tile (64 rows x 32 K bf16 = 4KB) staged in double-buffered LDS,
// shared by all 8 waves. A fragments register-double-buffered from global.
// Per K-step: batch 8x ds_load_b128 (all 4 B fragments) -> one wait ->
// 4 back-to-back WMMAs on independent accumulators.
// GATES=true maps LDS row r -> global B row (r>>4)*H + nbase + (r&15).
// =====================================================================
template <bool GATES>
__device__ __forceinline__ void gemm_core(const bf16* __restrict__ A, int lda, int mrow,
                                          const bf16* __restrict__ Bm, int ldb, int nbase,
                                          int K, bf16* smem, v8f acc[4]) {
  const int tid  = threadIdx.x;
  const int lane = tid & 31;
  const int half = lane >> 4, lm = lane & 15;

  // cooperative B-tile load: 256 threads x 16B = 4KB tile
  const int r    = tid >> 2;        // LDS row 0..63
  const int koff = (tid & 3) * 8;   // 0,8,16,24
  const int grow = GATES ? ((r >> 4) * HH + nbase + (r & 15)) : (nbase + r);
  const bf16* gsrc = Bm + (size_t)grow * ldb + koff;

  bf16* s0 = smem;
  bf16* s1 = smem + 64 * 32;

  // prologue: stage k=0 tile, load first A fragment
  v8bf breg = *(const v8bf*)(gsrc);
  *(v8bf*)(s0 + r * 32 + koff) = breg;
  __syncthreads();
  v16bf a_cur = load_a_frag(A, lda, mrow, 0, half);

  int buf = 0;
  for (int k = 0; k < K; k += 32) {
    const bool last = (k + 32 >= K);
    v8bf  bnext;
    v16bf a_next;
    if (!last) {                    // issue next-step global loads before computing
      bnext  = *(const v8bf*)(gsrc + (k + 32));
      a_next = load_a_frag(A, lda, mrow, k + 32, half);
    }
    bf16* sb = buf ? s1 : s0;
    v16bf bfrag[4];
#pragma unroll
    for (int i = 0; i < 4; ++i)     // batch all LDS reads: one clause, one wait
      bfrag[i] = *(const v16bf*)(sb + (i * 16 + lm) * 32 + half * 16);
#pragma unroll
    for (int i = 0; i < 4; ++i)     // 4 independent WMMAs, back-to-back
      acc[i] = WMMA_BF16(a_cur, bfrag[i], acc[i]);
    if (!last) {
      bf16* sw = buf ? s0 : s1;     // write opposite buffer (safe: last read 1 barrier ago)
      *(v8bf*)(sw + r * 32 + koff) = bnext;
      __syncthreads();
      buf ^= 1;
      a_cur = a_next;
    }
  }
}

// ---- elementwise prep kernels ----
__global__ void k_cvt(const float* __restrict__ s, bf16* __restrict__ d, int n) {
  int i = blockIdx.x * blockDim.x + threadIdx.x;
  if (i < n) d[i] = (bf16)s[i];
}

__global__ void k_cvt_pad(const float* __restrict__ s, bf16* __restrict__ d,
                          int rows, int padrows, int cols) {
  int i = blockIdx.x * blockDim.x + threadIdx.x;
  if (i >= padrows * cols) return;
  int r = i / cols;
  d[i] = (r < rows) ? (bf16)s[i] : (bf16)0.0f;
}

// shifted embeddings: A2[b*T+t, :] = emb[labels[b, (t+T-1)%T], :]
__global__ void k_gather(const int* __restrict__ labels, const float* __restrict__ emb,
                         bf16* __restrict__ A2) {
  int i = blockIdx.x * blockDim.x + threadIdx.x;
  if (i >= BB * TT * EE) return;
  int e = i % EE;
  int r = i / EE;
  int b = r / TT;
  int t = r % TT;
  int tt = (t + TT - 1) % TT;
  int lbl = labels[b * TT + tt];
  A2[i] = (bf16)emb[(size_t)lbl * EE + e];
}

__global__ void k_zero_hc(bf16* __restrict__ h0, float* __restrict__ c0, int n) {
  int i = blockIdx.x * blockDim.x + threadIdx.x;
  if (i < n) { h0[i] = (bf16)0.0f; c0[i] = 0.0f; }
}

// ---- generic WMMA GEMM: Out[M,N] = A[M,K] * W[N,K]^T (+bias +bias2) ----
template <bool OUT_BF16>
__global__ void k_gemm(const bf16* __restrict__ A, int lda,
                       const bf16* __restrict__ Bm, int ldb,
                       const float* __restrict__ bias, const float* __restrict__ bias2,
                       void* __restrict__ Out, int ldo, int Nstrips, int K) {
  __shared__ bf16 smem[2 * 64 * 32];
  int mb = blockIdx.x / Nstrips, ns = blockIdx.x % Nstrips;
  int w = threadIdx.x >> 5;
  int lane = threadIdx.x & 31, half = lane >> 4, lm = lane & 15;
  int mbase = mb * 128 + w * 16;
  int nbase = ns * 64;

  v8f acc[4] = {};
  gemm_core<false>(A, lda, mbase + lm, Bm, ldb, nbase, K, smem, acc);

#pragma unroll
  for (int i = 0; i < 4; ++i) {
    int n = nbase + i * 16 + lm;
    float bv = (bias ? bias[n] : 0.0f) + (bias2 ? bias2[n] : 0.0f);
#pragma unroll
    for (int r = 0; r < 8; ++r) {
      int m = mbase + r + 8 * half;
      float v = acc[i][r] + bv;
      if (OUT_BF16) ((bf16*)Out)[(size_t)m * ldo + n] = (bf16)v;
      else          ((float*)Out)[(size_t)m * ldo + n] = v;
    }
  }
}

// ---- LSTM step: g = Gemb[b*T+t] + Gfeat[b] + h_prev @ W_hh^T, fused cell update ----
// wave: 16 batch rows x 16 H-cols; its 4 tiles are the i/f/g/o gates at n+g*H.
__global__ void k_lstm_step(const bf16* __restrict__ Hprev, const bf16* __restrict__ Whhb,
                            const bf16* __restrict__ Gemb, const float* __restrict__ Gfeat,
                            float* __restrict__ Cbuf, bf16* __restrict__ Hnext,
                            bf16* __restrict__ HS, int t) {
  __shared__ bf16 smem[2 * 64 * 32];
  int mb = blockIdx.x / 64, ht = blockIdx.x % 64;   // 2 M-blocks x 64 H-tiles
  int w = threadIdx.x >> 5;
  int lane = threadIdx.x & 31, half = lane >> 4, lm = lane & 15;
  int mbase = mb * 128 + w * 16;
  int nbase = ht * 16;

  v8f acc[4] = {};
  gemm_core<true>(Hprev, HH, mbase + lm, Whhb, HH, nbase, HH, smem, acc);

  int n = nbase + lm;
#pragma unroll
  for (int r = 0; r < 8; ++r) {
    int m = mbase + r + 8 * half;
    size_t grow = ((size_t)m * TT + t) * (4 * HH);
    size_t frow = (size_t)m * (4 * HH);
    float gi = acc[0][r] + (float)Gemb[grow + 0 * HH + n] + Gfeat[frow + 0 * HH + n];
    float gf = acc[1][r] + (float)Gemb[grow + 1 * HH + n] + Gfeat[frow + 1 * HH + n];
    float gg = acc[2][r] + (float)Gemb[grow + 2 * HH + n] + Gfeat[frow + 2 * HH + n];
    float go = acc[3][r] + (float)Gemb[grow + 3 * HH + n] + Gfeat[frow + 3 * HH + n];
    float c  = sigmoidf(gf) * Cbuf[(size_t)m * HH + n] + sigmoidf(gi) * tanhf(gg);
    float h  = sigmoidf(go) * tanhf(c);
    Cbuf[(size_t)m * HH + n]          = c;
    Hnext[(size_t)m * HH + n]         = (bf16)h;
    HS[((size_t)t * BB + m) * HH + n] = (bf16)h;
  }
}

// ---- logits: out[(b*T+t)*V + n] = HS[t*B+b, :] . W_out[n, :] + b_out[n] ----
__global__ void k_logits(const bf16* __restrict__ HS, const bf16* __restrict__ Wob,
                         const float* __restrict__ b_out, float* __restrict__ out,
                         int Nstrips) {
  __shared__ bf16 smem[2 * 64 * 32];
  int mb = blockIdx.x / Nstrips, ns = blockIdx.x % Nstrips;
  int w = threadIdx.x >> 5;
  int lane = threadIdx.x & 31, half = lane >> 4, lm = lane & 15;
  int mbase = mb * 128 + w * 16;
  int nbase = ns * 64;

  v8f acc[4] = {};
  gemm_core<false>(HS, HH, mbase + lm, Wob, HH, nbase, HH, smem, acc);

#pragma unroll
  for (int i = 0; i < 4; ++i) {
    int n = nbase + i * 16 + lm;
    if (n < VV) {
      float bv = b_out[n];
#pragma unroll
      for (int r = 0; r < 8; ++r) {
        int m = mbase + r + 8 * half;     // m = t*B + b
        int t = m / BB, b = m % BB;
        out[((size_t)b * TT + t) * VV + n] = acc[i][r] + bv;
      }
    }
  }
}

extern "C" void kernel_launch(void* const* d_in, const int* in_sizes, int n_in,
                              void* d_out, int out_size, void* d_ws, size_t ws_size,
                              hipStream_t stream) {
  const float* X      = (const float*)d_in[0];
  const int*   labels = (const int*)  d_in[1];
  const float* W_f    = (const float*)d_in[2];
  const float* b_f    = (const float*)d_in[3];
  const float* emb    = (const float*)d_in[4];
  const float* W_ih   = (const float*)d_in[5];
  const float* W_hh   = (const float*)d_in[6];
  const float* b_ih   = (const float*)d_in[7];
  const float* b_hh   = (const float*)d_in[8];
  const float* W_out  = (const float*)d_in[9];
  const float* b_out  = (const float*)d_in[10];
  float* out = (float*)d_out;

  char* ws = (char*)d_ws;
  size_t off = 0;
  auto alloc = [&](size_t bytes) -> char* {
    char* p = ws + off;
    off = (off + bytes + 255) & ~(size_t)255;
    return p;
  };

  bf16*  Xb    = (bf16*) alloc((size_t)BB * INN * 2);
  bf16*  Wfb   = (bf16*) alloc((size_t)EE * INN * 2);
  bf16*  featb = (bf16*) alloc((size_t)BB * EE * 2);
  bf16*  Wihb  = (bf16*) alloc((size_t)4 * HH * 2 * EE * 2);
  bf16*  Whhb  = (bf16*) alloc((size_t)4 * HH * HH * 2);
  bf16*  Woutb = (bf16*) alloc((size_t)NPAD * HH * 2);
  bf16*  A2    = (bf16*) alloc((size_t)BB * TT * EE * 2);
  bf16*  Gemb  = (bf16*) alloc((size_t)BB * TT * 4 * HH * 2);
  float* Gfeat = (float*)alloc((size_t)BB * 4 * HH * 4);
  bf16*  Hb0   = (bf16*) alloc((size_t)BB * HH * 2);
  bf16*  Hb1   = (bf16*) alloc((size_t)BB * HH * 2);
  float* Cb    = (float*)alloc((size_t)BB * HH * 4);
  bf16*  HS    = (bf16*) alloc((size_t)TT * BB * HH * 2);

  auto cvtN = [&](const float* s, bf16* d, int n) {
    k_cvt<<<(n + 255) / 256, 256, 0, stream>>>(s, d, n);
  };
  cvtN(X,    Xb,   BB * INN);
  cvtN(W_f,  Wfb,  EE * INN);
  cvtN(W_ih, Wihb, 4 * HH * 2 * EE);
  cvtN(W_hh, Whhb, 4 * HH * HH);
  { int n = NPAD * HH;
    k_cvt_pad<<<(n + 255) / 256, 256, 0, stream>>>(W_out, Woutb, VV, NPAD, HH); }
  { int n = BB * TT * EE;
    k_gather<<<(n + 255) / 256, 256, 0, stream>>>(labels, emb, A2); }
  { int n = BB * HH;
    k_zero_hc<<<(n + 255) / 256, 256, 0, stream>>>(Hb0, Cb, n); }

  // features = X @ W_f^T + b_f : M=256, N=512, K=2048 -> bf16
  k_gemm<true><<<(BB / 128) * (EE / 64), 256, 0, stream>>>(
      Xb, INN, Wfb, INN, b_f, nullptr, featb, EE, EE / 64, INN);

  // gates_feat = features @ W_ih[:, :E]^T + b_ih + b_hh : M=256, N=4096, K=512 -> f32
  k_gemm<false><<<(BB / 128) * (4 * HH / 64), 256, 0, stream>>>(
      featb, EE, Wihb, 2 * EE, b_ih, b_hh, Gfeat, 4 * HH, 4 * HH / 64, EE);

  // gates_emb = shifted @ W_ih[:, E:]^T : M=10240, N=4096, K=512 -> bf16
  k_gemm<true><<<(BB * TT / 128) * (4 * HH / 64), 256, 0, stream>>>(
      A2, EE, Wihb + EE, 2 * EE, nullptr, nullptr, Gemb, 4 * HH, 4 * HH / 64, EE);

  // LSTM recurrence: 40 serial steps, (2 M-blocks x 64 H-tiles) = 128 WGs each
  for (int t = 0; t < TT; ++t) {
    const bf16* hp = (t & 1) ? Hb1 : Hb0;
    bf16*       hn = (t & 1) ? Hb0 : Hb1;
    k_lstm_step<<<(BB / 128) * (HH / 16), 256, 0, stream>>>(
        hp, Whhb, Gemb, Gfeat, Cb, hn, HS, t);
  }

  // logits: M=10240, Npad=5056, K=1024
  k_logits<<<(BB * TT / 128) * (NPAD / 64), 256, 0, stream>>>(
      HS, Woutb, b_out, out, NPAD / 64);
}
